// VQ_Frame_8821862826422
// MI455X (gfx1250) — matrix operand order repaired and checked by reading, not compiled, not run on hardware
//
#include <hip/hip_runtime.h>
#include <hip/hip_bf16.h>
#include <math.h>

#define N_ 32768
#define M_ 1024
#define D_ 512
#define K_ 4096

typedef _Float16 f16;
typedef __attribute__((ext_vector_type(16))) _Float16 v16h;
typedef __attribute__((ext_vector_type(8)))  _Float16 v8h;
typedef __attribute__((ext_vector_type(4)))  _Float16 v4h;
typedef __attribute__((ext_vector_type(8)))  float    v8f;

union AFrag { v16h v; v8h h[2]; };

// ---------------------------------------------------------------------------
// Split f32 -> (hi, lo) f16 pair: v = hi + lo to ~22 mantissa bits.
// ---------------------------------------------------------------------------
__global__ void k_split(const float* __restrict__ in, f16* __restrict__ oh,
                        f16* __restrict__ ol, int n) {
    int i = blockIdx.x * 256 + threadIdx.x;
    if (i < n) {
        float v = in[i];
        f16 h = (f16)v;
        oh[i] = h;
        ol[i] = (f16)(v - (float)h);
    }
}

// ---------------------------------------------------------------------------
// Codebook squared norms (exact f32): one wave per code row.
// ---------------------------------------------------------------------------
__global__ void k_cbnorm(const float* __restrict__ cb, float* __restrict__ cbn) {
    int lane = threadIdx.x;
    int k = blockIdx.x * blockDim.y + threadIdx.y;
    const float* row = cb + (size_t)k * D_;
    float s = 0.0f;
    for (int d = lane; d < D_; d += 32) {
        float v = row[d];
        s += v * v;
    }
    for (int m = 16; m > 0; m >>= 1) s += __shfl_xor(s, m, 32);
    if (lane == 0) cbn[k] = s;
}

// ---------------------------------------------------------------------------
// Encoder GEMM (split-f16 WMMA, K=32 steps) + fused BatchNorm.
// One wave -> one 16x16 tile of z.  x is split on the fly (VALU co-executes
// with the XDL wmma pipe); enc_w comes pre-split from workspace.
// f16 A layout: lanes 0-15 K=kk..kk+15; lanes 16-31 K=kk+16..kk+31 (contiguous, same as B)
// f16 B layout: 16 contiguous f16 at column row, offset kk + 16*half.
// ---------------------------------------------------------------------------
__global__ void k_encoder(const float* __restrict__ x,
                          const f16* __restrict__ wh,
                          const f16* __restrict__ wl,
                          const float* __restrict__ enc_b,
                          const float* __restrict__ gamma,
                          const float* __restrict__ beta,
                          const float* __restrict__ mean,
                          const float* __restrict__ var,
                          float* __restrict__ z) {
    int lane = threadIdx.x;
    int hh = lane >> 4;
    int lr = lane & 15;
    int row0 = blockIdx.x * 16;
    int col0 = (blockIdx.y * blockDim.y + threadIdx.y) * 16;

    const float* xrow = x + (size_t)(row0 + lr) * M_;
    const f16* bhrow = wh + (size_t)(col0 + lr) * M_;
    const f16* blrow = wl + (size_t)(col0 + lr) * M_;

    v8f acc = {};
    for (int kk = 0; kk < M_; kk += 32) {
        const float* p0 = xrow + kk + 16 * hh;       // K chunk 0
        const float* p1 = p0 + 8;                    // K chunk 1
        float4 q0 = *(const float4*)(p0);
        float4 q1 = *(const float4*)(p0 + 4);
        float4 q2 = *(const float4*)(p1);
        float4 q3 = *(const float4*)(p1 + 4);
        float fs[16] = {q0.x, q0.y, q0.z, q0.w, q1.x, q1.y, q1.z, q1.w,
                        q2.x, q2.y, q2.z, q2.w, q3.x, q3.y, q3.z, q3.w};
        v16h ah, al;
#pragma unroll
        for (int j = 0; j < 16; j++) {
            f16 h = (f16)fs[j];
            ah[j] = h;
            al[j] = (f16)(fs[j] - (float)h);
        }
        v16h bh = *(const v16h*)(bhrow + kk + 16 * hh);
        v16h bl = *(const v16h*)(blrow + kk + 16 * hh);
        acc = __builtin_amdgcn_wmma_f32_16x16x32_f16(false, ah, false, bh, (short)0, acc, false, false);
        acc = __builtin_amdgcn_wmma_f32_16x16x32_f16(false, ah, false, bl, (short)0, acc, false, false);
        acc = __builtin_amdgcn_wmma_f32_16x16x32_f16(false, al, false, bh, (short)0, acc, false, false);
    }

    int d = col0 + lr;
    float s = rsqrtf(var[d] + 1e-5f) * gamma[d];
    float sh = (enc_b[d] - mean[d]) * s + beta[d];
#pragma unroll
    for (int i = 0; i < 8; i++) {
        int r = row0 + i + 8 * hh;
        z[(size_t)r * D_ + d] = acc[i] * s + sh;
    }
}

// ---------------------------------------------------------------------------
// VQ nearest-code search, split-f16 WMMA distance GEMM, big-LDS tiling.
// Block = 8 waves x 16 rows = 128 rows.  Each wave keeps its own split z tile
// in LDS; the 16-code codebook tile (hi+lo) is staged cooperatively and shared
// by all 8 waves -> codebook L2 traffic drops 8x (16 GB -> 2 GB).
// Total dynamic LDS ~293 KB (CDNA5 allows 320 KB per workgroup).
// dist(row, code) = ||e||^2 - 2 * z.e   (||z||^2 constant per row, dropped)
// ---------------------------------------------------------------------------
#define ZSTR 520
#define VQ_WAVES 8

__global__ void k_vq(const float* __restrict__ z,
                     const f16* __restrict__ cbh,
                     const f16* __restrict__ cbl,
                     const float* __restrict__ cbn,
                     int* __restrict__ idxb) {
    extern __shared__ __align__(16) f16 smem[];
    f16* zhS = smem;                                  // [VQ_WAVES*16][ZSTR]
    f16* zlS = zhS + VQ_WAVES * 16 * ZSTR;            // [VQ_WAVES*16][ZSTR]
    f16* chS = zlS + VQ_WAVES * 16 * ZSTR;            // [16][ZSTR]
    f16* clS = chS + 16 * ZSTR;                       // [16][ZSTR]

    int lane = threadIdx.x;
    int ty = threadIdx.y;
    int tid = ty * 32 + lane;
    int row0 = blockIdx.x * (VQ_WAVES * 16);
    int rbase = ty * 16;                              // this wave's LDS tile

    // Per-wave stage + split of its own 16x512 z tile.
    for (int t = lane; t < 16 * 128; t += 32) {
        int r = t >> 7;
        int c = (t & 127) << 2;
        float4 q = *(const float4*)&z[(size_t)(row0 + rbase + r) * D_ + c];
        float fs[4] = {q.x, q.y, q.z, q.w};
        v4h H, L;
#pragma unroll
        for (int j = 0; j < 4; j++) {
            f16 h = (f16)fs[j];
            H[j] = h;
            L[j] = (f16)(fs[j] - (float)h);
        }
        *(v4h*)&zhS[(rbase + r) * ZSTR + c] = H;
        *(v4h*)&zlS[(rbase + r) * ZSTR + c] = L;
    }
    __syncthreads();

    int hh = lane >> 4;
    int lr = lane & 15;

    float best[8];
    int bidx[8];
#pragma unroll
    for (int i = 0; i < 8; i++) { best[i] = 3.4e38f; bidx[i] = 0; }

    for (int ct = 0; ct < K_; ct += 16) {
        // Cooperative stage of this codebook tile (16 codes x 512, hi+lo).
        for (int t = tid; t < 16 * 64; t += 32 * VQ_WAVES) {
            int r = t >> 6;
            int c = (t & 63) << 3;
            *(v8h*)&chS[r * ZSTR + c] = *(const v8h*)&cbh[(size_t)(ct + r) * D_ + c];
            *(v8h*)&clS[r * ZSTR + c] = *(const v8h*)&cbl[(size_t)(ct + r) * D_ + c];
        }
        __syncthreads();

        v8f acc = {};
        for (int kk = 0; kk < D_; kk += 32) {
            int a0 = (rbase + lr) * ZSTR + kk + 16 * hh;
            AFrag ah, al;
            ah.h[0] = *(const v8h*)&zhS[a0];
            ah.h[1] = *(const v8h*)&zhS[a0 + 8];
            al.h[0] = *(const v8h*)&zlS[a0];
            al.h[1] = *(const v8h*)&zlS[a0 + 8];
            int b0 = lr * ZSTR + kk + 16 * hh;
            AFrag bh, bl;
            bh.h[0] = *(const v8h*)&chS[b0];
            bh.h[1] = *(const v8h*)&chS[b0 + 8];
            bl.h[0] = *(const v8h*)&clS[b0];
            bl.h[1] = *(const v8h*)&clS[b0 + 8];
            acc = __builtin_amdgcn_wmma_f32_16x16x32_f16(false, ah.v, false, bh.v, (short)0, acc, false, false);
            acc = __builtin_amdgcn_wmma_f32_16x16x32_f16(false, ah.v, false, bl.v, (short)0, acc, false, false);
            acc = __builtin_amdgcn_wmma_f32_16x16x32_f16(false, al.v, false, bh.v, (short)0, acc, false, false);
        }
        int code = ct + lr;
        float cn = cbn[code];
#pragma unroll
        for (int i = 0; i < 8; i++) {
            float dist = cn - 2.0f * acc[i];
            if (dist < best[i]) { best[i] = dist; bidx[i] = code; }
        }
        __syncthreads();   // protect chS/clS before next stage
    }

    // Each wave saw all codes for its own rows: pure in-wave argmin reduce
    // across the 16 lanes of each half (tie -> lower code index).
#pragma unroll
    for (int i = 0; i < 8; i++) {
        float v = best[i];
        int id = bidx[i];
        for (int m = 1; m < 16; m <<= 1) {
            float ov = __shfl_xor(v, m, 32);
            int oid = __shfl_xor(id, m, 32);
            if (ov < v || (ov == v && oid < id)) { v = ov; id = oid; }
        }
        if (lr == 0) idxb[row0 + rbase + i + 8 * hh] = id;
    }
}

// ---------------------------------------------------------------------------
// One wave per row: one-hot scatter, histogram, commitment-loss partial (f32).
// ---------------------------------------------------------------------------
__global__ void k_quant_stats(const float* __restrict__ z,
                              const float* __restrict__ cb,
                              const int* __restrict__ idxb,
                              float* __restrict__ enc,
                              unsigned* __restrict__ counts,
                              float* __restrict__ lossacc) {
    int lane = threadIdx.x;
    int n = blockIdx.x * blockDim.y + threadIdx.y;
    int id = idxb[n];
    const float* q = cb + (size_t)id * D_;
    const float* zr = z + (size_t)n * D_;
    float s = 0.0f;
    for (int d = lane; d < D_; d += 32) {
        float df = q[d] - zr[d];
        s += df * df;
    }
    for (int m = 16; m > 0; m >>= 1) s += __shfl_xor(s, m, 32);
    if (lane == 0) {
        atomicAdd(lossacc, s);
        atomicAdd(&counts[id], 1u);
        enc[(size_t)n * K_ + id] = 1.0f;
    }
}

// ---------------------------------------------------------------------------
// Decoder GEMM (split-f16 WMMA) with A rows gathered from pre-split codebook:
// out = codebook[idx] @ dec_w^T + dec_b
// ---------------------------------------------------------------------------
__global__ void k_decoder(const f16* __restrict__ cbh,
                          const f16* __restrict__ cbl,
                          const int* __restrict__ idxb,
                          const f16* __restrict__ dwh,
                          const f16* __restrict__ dwl,
                          const float* __restrict__ dec_b,
                          float* __restrict__ out) {
    int lane = threadIdx.x;
    int hh = lane >> 4;
    int lr = lane & 15;
    int row0 = blockIdx.x * 16;
    int col0 = (blockIdx.y * blockDim.y + threadIdx.y) * 16;

    int id = idxb[row0 + lr];
    const f16* ahrow = cbh + (size_t)id * D_;
    const f16* alrow = cbl + (size_t)id * D_;
    const f16* bhrow = dwh + (size_t)(col0 + lr) * D_;
    const f16* blrow = dwl + (size_t)(col0 + lr) * D_;

    v8f acc = {};
    for (int kk = 0; kk < D_; kk += 32) {
        int a0 = kk + 16 * hh;
        AFrag ah, al;
        ah.h[0] = *(const v8h*)(ahrow + a0);
        ah.h[1] = *(const v8h*)(ahrow + a0 + 8);
        al.h[0] = *(const v8h*)(alrow + a0);
        al.h[1] = *(const v8h*)(alrow + a0 + 8);
        v16h bh = *(const v16h*)(bhrow + kk + 16 * hh);
        v16h bl = *(const v16h*)(blrow + kk + 16 * hh);
        acc = __builtin_amdgcn_wmma_f32_16x16x32_f16(false, ah.v, false, bh, (short)0, acc, false, false);
        acc = __builtin_amdgcn_wmma_f32_16x16x32_f16(false, ah.v, false, bl, (short)0, acc, false, false);
        acc = __builtin_amdgcn_wmma_f32_16x16x32_f16(false, al.v, false, bh, (short)0, acc, false, false);
    }

    int m = col0 + lr;
    float bb = dec_b[m];
#pragma unroll
    for (int i = 0; i < 8; i++) {
        int r = row0 + i + 8 * hh;
        out[(size_t)r * M_ + m] = acc[i] + bb;
    }
}

// ---------------------------------------------------------------------------
// Scalars: perplexity from counts, loss from accumulator.
// ---------------------------------------------------------------------------
__global__ void k_finalize(const unsigned* __restrict__ counts,
                           const float* __restrict__ lossacc,
                           float* __restrict__ loss_out,
                           float* __restrict__ perp_out) {
    __shared__ float red[256];
    int tid = threadIdx.x;
    float e = 0.0f;
    for (int k = tid; k < K_; k += 256) {
        float p = (float)counts[k] * (1.0f / (float)N_);
        e -= p * logf(p + 1e-10f);
    }
    red[tid] = e;
    __syncthreads();
    for (int s = 128; s > 0; s >>= 1) {
        if (tid < s) red[tid] += red[tid + s];
        __syncthreads();
    }
    if (tid == 0) {
        *perp_out = expf(red[0]);
        *loss_out = 0.25f * (*lossacc) / ((float)N_ * (float)D_);
    }
}

extern "C" void kernel_launch(void* const* d_in, const int* in_sizes, int n_in,
                              void* d_out, int out_size, void* d_ws, size_t ws_size,
                              hipStream_t stream) {
    const float* x       = (const float*)d_in[0];
    const float* enc_w   = (const float*)d_in[1];
    const float* enc_b   = (const float*)d_in[2];
    const float* bn_g    = (const float*)d_in[3];
    const float* bn_b    = (const float*)d_in[4];
    const float* bn_mu   = (const float*)d_in[5];
    const float* bn_var  = (const float*)d_in[6];
    const float* cbook   = (const float*)d_in[7];
    const float* dec_w   = (const float*)d_in[8];
    const float* dec_b   = (const float*)d_in[9];

    float* out      = (float*)d_out;
    float* loss_ptr = out + (size_t)N_ * M_;
    float* perp_ptr = loss_ptr + 1;
    float* enc      = perp_ptr + 1;                 // [N, K] one-hot

    // Workspace layout (f16 region 32B-aligned for v16h loads)
    float*    ws      = (float*)d_ws;
    float*    z       = ws;                         // N*D f32
    float*    cbn     = z + (size_t)N_ * D_;        // K
    unsigned* counts  = (unsigned*)(cbn + K_);      // K
    int*      idxb    = (int*)(counts + K_);        // N
    float*    lossacc = (float*)(idxb + N_);        // 1 (+7 pad for alignment)
    f16*      f16base = (f16*)(lossacc + 8);
    f16* enc_wh = f16base;                          // D*M
    f16* enc_wl = enc_wh + (size_t)D_ * M_;
    f16* dec_wh = enc_wl + (size_t)D_ * M_;         // M*D
    f16* dec_wl = dec_wh + (size_t)M_ * D_;
    f16* cb_h   = dec_wl + (size_t)M_ * D_;         // K*D
    f16* cb_l   = cb_h + (size_t)K_ * D_;

    // Zero one-hot output region and accumulators (every call).
    hipMemsetAsync(enc, 0, (size_t)N_ * K_ * sizeof(float), stream);
    hipMemsetAsync(counts, 0, K_ * sizeof(unsigned), stream);
    hipMemsetAsync(lossacc, 0, sizeof(float), stream);

    // Pre-split the small, heavily-reused matrices into f16 hi/lo.
    int nw = D_ * M_;
    k_split<<<dim3((nw + 255) / 256), dim3(256), 0, stream>>>(enc_w, enc_wh, enc_wl, nw);
    k_split<<<dim3((nw + 255) / 256), dim3(256), 0, stream>>>(dec_w, dec_wh, dec_wl, nw);
    int nc = K_ * D_;
    k_split<<<dim3((nc + 255) / 256), dim3(256), 0, stream>>>(cbook, cb_h, cb_l, nc);

    k_cbnorm<<<dim3(K_ / 8), dim3(32, 8), 0, stream>>>(cbook, cbn);

    k_encoder<<<dim3(N_ / 16, D_ / 64), dim3(32, 4), 0, stream>>>(
        x, enc_wh, enc_wl, enc_b, bn_g, bn_b, bn_mu, bn_var, z);

    // Dynamic LDS: 2 * (8*16 + 16) rows * ZSTR f16 = ~293 KB (<= 320 KB/WGP).
    size_t vq_lds = (size_t)(2 * VQ_WAVES * 16 * ZSTR + 2 * 16 * ZSTR) * sizeof(f16);
    k_vq<<<dim3(N_ / (VQ_WAVES * 16)), dim3(32, VQ_WAVES), vq_lds, stream>>>(
        z, cb_h, cb_l, cbn, idxb);

    k_quant_stats<<<dim3(N_ / 8), dim3(32, 8), 0, stream>>>(
        z, cbook, idxb, enc, counts, lossacc);

    k_decoder<<<dim3(N_ / 16, M_ / 64), dim3(32, 4), 0, stream>>>(
        cb_h, cb_l, idxb, dec_wh, dec_wl, dec_b, out);

    k_finalize<<<1, 256, 0, stream>>>(counts, lossacc, loss_ptr, perp_ptr);
}
